// FusionModel_49589692400276
// MI455X (gfx1250) — 16-bit and 8-bit matrix operand order repaired and checked by reading, compiled, not run on hardware
//
#include <hip/hip_runtime.h>
#include <hip/hip_bf16.h>
#include <stdint.h>

#define B_  16
#define T_  4096
#define D_  256
#define H_  256
#define G4_ 1024

typedef __attribute__((ext_vector_type(8)))  float  v8f;
typedef __attribute__((ext_vector_type(16))) int    v16i;
typedef __attribute__((ext_vector_type(16))) __bf16 v16bf;


__device__ __forceinline__ unsigned short f2bf(float f) {
  union { float f; unsigned u; } x; x.f = f;
  unsigned r = x.u + 0x7fffu + ((x.u >> 16) & 1u);
  return (unsigned short)(r >> 16);
}

__device__ __forceinline__ unsigned char f2e4m3(float f) {
  union { float f; unsigned u; } x; x.f = f;
  unsigned s = (x.u >> 24) & 0x80u;
  float ax = __builtin_fabsf(f);
  if (ax > 448.f) ax = 448.f;
  x.f = ax;
  int e = (int)((x.u >> 23) & 0xffu) - 127;
  unsigned m = x.u & 0x7fffffu;
  if (e >= -6) {
    unsigned mant = (m >> 20) + ((m >> 19) & 1u);
    unsigned ee = (unsigned)(e + 7);
    if (mant >= 8u) { mant = 0u; ee += 1u; }
    if (ee > 15u || (ee == 15u && mant >= 7u)) { ee = 15u; mant = 6u; }
    return (unsigned char)(s | (ee << 3) | mant);
  } else if (e >= -9) {
    unsigned sig = 0x800000u | m;
    int sh = 14 - e;
    unsigned q = (sig >> sh) + ((sig >> (sh - 1)) & 1u);
    return (unsigned char)(s | q);
  }
  return (unsigned char)s;
}

__device__ __forceinline__ unsigned char f2e4m3_s(float v) {
  union { float f; unsigned u; } x; x.f = v;
  unsigned s  = (x.u >> 24) & 0x80u;
  unsigned ef = (x.u >> 23) & 0xffu;
  unsigned m  = x.u & 0x7fffffu;
  unsigned mant = (m >> 20) + ((m >> 19) & 1u);
  unsigned ee = ef - 120u + (mant >> 3);
  mant &= 7u;
  unsigned byte = s | (ee << 3) | mant;
  return (unsigned char)((ef < 121u) ? s : byte);
}

__device__ __forceinline__ v8f v8f_zero() {
  v8f z;
  #pragma unroll
  for (int i = 0; i < 8; ++i) z[i] = 0.f;
  return z;
}

__device__ __forceinline__ float sigf(float x) { return 1.f / (1.f + __expf(-x)); }
__device__ __forceinline__ float tanhfast(float x) { return 1.f - 2.f / (__expf(2.f * x) + 1.f); }


__global__ __launch_bounds__(256) void cvt_bf16(const float* __restrict__ x,
                                                const float* __restrict__ Wih,
                                                unsigned short* __restrict__ xbf,
                                                unsigned short* __restrict__ wbf) {
  const size_t NX = (size_t)B_ * T_ * D_;
  size_t base = ((size_t)blockIdx.x * blockDim.x + threadIdx.x) * 8;
  const float* src;
  unsigned short* dst;
  if (base < NX) { src = x + base; dst = xbf + base; }
  else           { size_t o = base - NX; src = Wih + o; dst = wbf + o; }
  float4 f0 = *(const float4*)(src);
  float4 f1 = *(const float4*)(src + 4);
  union { unsigned short u[8]; int4 q; } h;
  h.u[0] = f2bf(f0.x); h.u[1] = f2bf(f0.y); h.u[2] = f2bf(f0.z); h.u[3] = f2bf(f0.w);
  h.u[4] = f2bf(f1.x); h.u[5] = f2bf(f1.y); h.u[6] = f2bf(f1.z); h.u[7] = f2bf(f1.w);
  *(int4*)dst = h.q;
}


__global__ __launch_bounds__(256) void quant_whh(const float* __restrict__ Whh,
                                                 unsigned char* __restrict__ Wq,
                                                 float* __restrict__ scl) {
  int row  = (blockIdx.x * blockDim.x + threadIdx.x) >> 5;
  int lane = threadIdx.x & 31;
  const float* rp = Whh + (size_t)row * H_;
  float v[8]; float m = 0.f;
  #pragma unroll
  for (int i = 0; i < 8; ++i) { v[i] = rp[lane * 8 + i]; m = fmaxf(m, __builtin_fabsf(v[i])); }
  #pragma unroll
  for (int off = 16; off > 0; off >>= 1) m = fmaxf(m, __shfl_xor(m, off, 32));
  float scale = fmaxf(m, 1e-20f) * (1.0f / 448.0f);
  float inv = 1.0f / scale;
  unsigned char q[8];
  #pragma unroll
  for (int i = 0; i < 8; ++i) q[i] = f2e4m3(v[i] * inv);
  uint2 p;
  p.x = (unsigned)q[0] | ((unsigned)q[1] << 8) | ((unsigned)q[2] << 16) | ((unsigned)q[3] << 24);
  p.y = (unsigned)q[4] | ((unsigned)q[5] << 8) | ((unsigned)q[6] << 16) | ((unsigned)q[7] << 24);
  *(uint2*)(Wq + (size_t)row * H_ + lane * 8) = p;
  if (lane == 0) scl[row] = scale;
}


#define XSTR 528

__global__ __launch_bounds__(256) void xproj_gemm(const unsigned short* __restrict__ xbf,
                                                  const unsigned short* __restrict__ wbf,
                                                  const float* __restrict__ bih,
                                                  const float* __restrict__ bhh,
                                                  float* __restrict__ xpo) {
  __shared__ unsigned char lA[16 * XSTR];
  const int mtile = blockIdx.x >> 1;
  const int nhalf = (blockIdx.x & 1) * 512;
  const int tid = threadIdx.x;

  {
    int row = tid >> 4, seg = tid & 15;
    const unsigned short* src = xbf + (size_t)(mtile * 16 + row) * D_ + seg * 16;
    *(int4*)(lA + row * XSTR + seg * 32)      = *(const int4*)src;
    *(int4*)(lA + row * XSTR + seg * 32 + 16) = *(const int4*)(src + 8);
  }
  __syncthreads();

  const int wave = tid >> 5, lane = tid & 31;
  const int lrow = lane & 15, lhi = lane >> 4;
  const int nbase = nhalf + wave * 64;

  v8f acc[4];
  #pragma unroll
  for (int nt = 0; nt < 4; ++nt) acc[nt] = v8f_zero();

  #pragma unroll
  for (int kk = 0; kk < 8; ++kk) {
    const int k0 = kk * 32;
    union { int4 q[2]; v16bf v; } af;
    const unsigned char* ap = lA + lrow * XSTR + (k0 + lhi * 8) * 2;
    af.q[0] = *(const int4*)ap;
    af.q[1] = *(const int4*)(ap + 32);
    #pragma unroll
    for (int nt = 0; nt < 4; ++nt) {
      const int n0 = nbase + nt * 16 + lrow;
      const unsigned short* bp = wbf + (size_t)n0 * D_ + k0 + lhi * 8;
      union { int4 q[2]; v16bf v; } bf;
      bf.q[0] = *(const int4*)(bp);
      bf.q[1] = *(const int4*)(bp + 16);
      acc[nt] = __builtin_amdgcn_wmma_f32_16x16x32_bf16(
          false, af.v, false, bf.v, (short)0, acc[nt], false, false);
    }
  }

  #pragma unroll
  for (int nt = 0; nt < 4; ++nt) {
    const int n0 = nbase + nt * 16 + lrow;
    const float bias = bih[n0] + bhh[n0];
    #pragma unroll
    for (int r = 0; r < 8; ++r) {
      const int mrow = mtile * 16 + r + lhi * 8;
      const int b = mrow >> 12;
      const int t = mrow & (T_ - 1);
      xpo[((size_t)t * B_ + b) * G4_ + n0] = acc[nt][r] + bias;
    }
  }
}


#define HSTR 272

__global__ __launch_bounds__(256) void lstm_scan(const float* __restrict__ xp,
                                                 const unsigned char* __restrict__ Wq,
                                                 const float* __restrict__ scl,
                                                 float* __restrict__ hout) {
  __shared__ unsigned char hl[16 * HSTR];
  const int tid = threadIdx.x;
  for (int i = tid; i < (16 * HSTR) / 4; i += 256) ((int*)hl)[i] = 0;

  const int wave = tid >> 5, lane = tid & 31;
  const int lrow = lane & 15, lhi = lane >> 4;

  v16i wB[8][2];
  float sc[8];
  int   ncol[8];
  #pragma unroll
  for (int g = 0; g < 4; ++g) {
    #pragma unroll
    for (int hf = 0; hf < 2; ++hf) {
      const int tt = g * 2 + hf;
      const int n  = g * 256 + wave * 32 + hf * 16 + lrow;
      ncol[tt] = n;
      sc[tt]   = scl[n] * (1.0f / 256.0f);
      const unsigned char* wp = Wq + (size_t)n * H_ + lhi * 16;
      #pragma unroll
      for (int kh = 0; kh < 2; ++kh) {
        union { int4 q[4]; v16i v; } u;
        #pragma unroll
        for (int c = 0; c < 4; ++c)
          u.q[c] = *(const int4*)(wp + kh * 128 + c * 32);
        wB[tt][kh] = u.v;
      }
    }
  }

  float cst[16];
  #pragma unroll
  for (int i = 0; i < 16; ++i) cst[i] = 0.f;

  const int hj0 = wave * 32 + lrow;
  __syncthreads();

  for (int t = 0; t < T_; ++t) {
    const size_t stepbase = (size_t)t * (B_ * G4_);
    float xv[8][8];
    #pragma unroll
    for (int tt = 0; tt < 8; ++tt) {
      #pragma unroll
      for (int r = 0; r < 8; ++r) {
        const int b = r + 8 * lhi;
        xv[tt][r] = xp[stepbase + (size_t)b * G4_ + ncol[tt]];
      }
    }
    {
      const int tn = (t + 1 < T_) ? (t + 1) : t;
      __builtin_prefetch(xp + (size_t)tn * (B_ * G4_) + (size_t)tid * 64, 0, 1);
    }

    union { int2 d[8]; v16i v; } a0, a1;
    const unsigned char* hp = hl + lrow * HSTR + lhi * 16;
    #pragma unroll
    for (int c = 0; c < 8; ++c) {
      a0.d[c] = *(const int2*)(hp + (c >> 1) * 32 + (c & 1) * 8);
      a1.d[c] = *(const int2*)(hp + 128 + (c >> 1) * 32 + (c & 1) * 8);
    }
    __syncthreads();

    v8f acc[8];
    #pragma unroll
    for (int tt = 0; tt < 8; ++tt) {
      v8f z = v8f_zero();
      z = __builtin_amdgcn_wmma_f32_16x16x128_fp8_fp8(a0.v, wB[tt][0], (short)0, z, false, false);
      acc[tt] = __builtin_amdgcn_wmma_f32_16x16x128_fp8_fp8(a1.v, wB[tt][1], (short)0, z, false, false);
    }

    #pragma unroll
    for (int hf = 0; hf < 2; ++hf) {
      const int j = hj0 + hf * 16;
      #pragma unroll
      for (int r = 0; r < 8; ++r) {
        const int b = r + 8 * lhi;
        const float gi = xv[0 + hf][r] + sc[0 + hf] * acc[0 + hf][r];
        const float gf = xv[2 + hf][r] + sc[2 + hf] * acc[2 + hf][r];
        const float gg = xv[4 + hf][r] + sc[4 + hf] * acc[4 + hf][r];
        const float go = xv[6 + hf][r] + sc[6 + hf] * acc[6 + hf][r];
        const float cc = sigf(gf) * cst[hf * 8 + r] + sigf(gi) * tanhfast(gg);
        cst[hf * 8 + r] = cc;
        const float h = sigf(go) * tanhfast(cc);
        hl[b * HSTR + j] = f2e4m3_s(h * 256.0f);
        hout[((size_t)b * T_ + t) * H_ + j] = h;
      }
    }
    __syncthreads();
  }
}


__global__ __launch_bounds__(256) void gate_logits(const float* __restrict__ hseq,
                                                   const float* __restrict__ Wlin,
                                                   const float* __restrict__ blin,
                                                   float* __restrict__ gbuf) {
  const int row = blockIdx.x * blockDim.x + threadIdx.x;
  const float* hp = hseq + (size_t)row * H_;
  float a0 = 0.f, a1 = 0.f;
  for (int k = 0; k < H_; k += 4) {
    float4 hv = *(const float4*)(hp + k);
    float4 w0 = *(const float4*)(Wlin + k);
    float4 w1 = *(const float4*)(Wlin + H_ + k);
    a0 += hv.x * w0.x + hv.y * w0.y + hv.z * w0.z + hv.w * w0.w;
    a1 += hv.x * w1.x + hv.y * w1.y + hv.z * w1.z + hv.w * w1.w;
  }
  a0 += blin[0]; a1 += blin[1];
  gbuf[row] = (a1 > a0) ? 1.0f : 0.0f;
}


__global__ __launch_bounds__(256) void scan_combine(const float* __restrict__ hseq,
                                                    const float* __restrict__ gbuf,
                                                    const float* __restrict__ fw,
                                                    const float* __restrict__ bw,
                                                    float* __restrict__ out) {
  const int b = blockIdx.x, j = threadIdx.x;
  const float fwv = fw[j], bwv = bw[j];
  const float* hp = hseq + (size_t)b * T_ * H_ + j;
  const float* gp = gbuf + (size_t)b * T_;
  float total = 0.f;
  for (int t = 0; t < T_; ++t) total += gp[t] * hp[(size_t)t * H_];
  float inc = 0.f;
  float* op = out + (size_t)b * T_ * H_ + j;
  for (int t = 0; t < T_; ++t) {
    const float l = hp[(size_t)t * H_];
    const float w = gp[t] * l;
    inc += w;
    op[(size_t)t * H_] = 2.f * l + (inc - w) * fwv + (total - inc) * bwv;
  }
}


extern "C" void kernel_launch(void* const* d_in, const int* in_sizes, int n_in,
                              void* d_out, int out_size, void* d_ws, size_t ws_size,
                              hipStream_t stream) {
  const float* x    = (const float*)d_in[0];
  const float* Wih  = (const float*)d_in[1];
  const float* Whh  = (const float*)d_in[2];
  const float* bih  = (const float*)d_in[3];
  const float* bhh  = (const float*)d_in[4];
  const float* Wlin = (const float*)d_in[5];
  const float* blin = (const float*)d_in[6];
  const float* fw   = (const float*)d_in[7];
  const float* bw   = (const float*)d_in[8];
  float* out = (float*)d_out;

  unsigned char* ws = (unsigned char*)d_ws;
  size_t off = 0;
  float*          xpo  = (float*)(ws + off);          off += 268435456ull;
  float*          hseq = (float*)(ws + off);          off += 67108864ull;
  unsigned char*  Wq   = ws + off;                    off += 262144ull;
  float*          scl  = (float*)(ws + off);          off += 4096ull;
  float*          gbuf = (float*)(ws + off);          off += 262144ull;
  unsigned short* xbf  = (unsigned short*)(ws + off); off += 33554432ull;
  unsigned short* wbf  = (unsigned short*)(ws + off); off += 524288ull;

  const int cvt_blocks = (int)(((size_t)B_ * T_ * D_ + 1024ull * 256ull) / 8ull / 256ull);
  cvt_bf16   <<<cvt_blocks, 256, 0, stream>>>(x, Wih, xbf, wbf);
  quant_whh  <<<128, 256, 0, stream>>>(Whh, Wq, scl);
  xproj_gemm <<<(B_ * T_ / 16) * 2, 256, 0, stream>>>(xbf, wbf, bih, bhh, xpo);
  lstm_scan  <<<1, 256, 0, stream>>>(xpo, Wq, scl, hseq);
  gate_logits<<<B_ * T_ / 256, 256, 0, stream>>>(hseq, Wlin, blin, gbuf);
  scan_combine<<<B_, 256, 0, stream>>>(hseq, gbuf, fw, bw, out);
}
